// DecoderKWTL_89180700934342
// MI455X (gfx1250) — matrix operand order repaired and checked by reading, not compiled, not run on hardware
//
#include <hip/hip_runtime.h>
#include <hip/hip_bf16.h>
#include <math.h>

#define B_     128
#define T_     30
#define NPIX_  49
#define ENC_   2048
#define HID_   512
#define EMB_   512
#define ATT_   512
#define VOCAB_ 10000
#define NGATE_ 2560   // 4*HID (lstm gates) + HID (sentinel gate)

typedef __bf16 bf16_t;
typedef __attribute__((ext_vector_type(16))) __bf16 v16bf;
typedef __attribute__((ext_vector_type(8)))  __bf16 v8bf;
typedef __attribute__((ext_vector_type(8)))  float  v8f;

__device__ __forceinline__ float sigmoidf_(float x) { return 1.0f / (1.0f + __expf(-x)); }

// ---------------------------------------------------------------------------
// WMMA helpers.  A fragment (contiguous k = 16*kh + s order) for lane L = two contiguous
// 8 x bf16 (16B) runs of row (lane&15):  [k0+kh*16, +8) and [k0+kh*16+8, +8).
// B fragment = 16 contiguous bf16 of a W row (32B vector load).
// ---------------------------------------------------------------------------
struct GemmSet {
    const bf16_t* X; int ldX;      // A source [M,K], bf16
    const bf16_t* W;               // [N,K] row-major bf16
    const float*  bias;            // nullable [N]
    float*        Y; int ldY;      // f32 out
    bf16_t*       Ybf;             // nullable bf16 copy, row stride = N
    const float*  Yin; int ldYin;  // nullable f32 accumulate-from
    int N, K;
    int act;                       // 0 none, 1 relu, 2 tanh
    const int* mlen; int tcur;     // nullable row mask: valid iff mlen[m] > tcur
};

__device__ __forceinline__ void gemm_tile(const GemmSet g, int tile, int lane)
{
    const int ntn   = g.N >> 4;
    const int tm    = tile / ntn;
    const int tn    = tile - tm * ntn;
    const int khalf = lane >> 4;
    const int mrow  = tm * 16 + (lane & 15);
    const int ncol  = tn * 16 + (lane & 15);

    const bf16_t* xrow = g.X + (size_t)mrow * g.ldX;
    const bf16_t* wrow = g.W + (size_t)ncol * g.K + khalf * 16;

    v8f acc = {};
    for (int k0 = 0; k0 < g.K; k0 += 32) {
        v8bf alo = *(const v8bf*)(xrow + k0 + khalf * 16);
        v8bf ahi = *(const v8bf*)(xrow + k0 + 8 + khalf * 16);
        v16bf a;
#pragma unroll
        for (int e = 0; e < 8; ++e) { a[e] = alo[e]; a[8 + e] = ahi[e]; }
        v16bf b = *(const v16bf*)(wrow + k0);
        acc = __builtin_amdgcn_wmma_f32_16x16x32_bf16(false, a, false, b,
                                                      (short)0, acc, false, false);
    }

    const float bv = g.bias ? g.bias[ncol] : 0.0f;
#pragma unroll
    for (int e = 0; e < 8; ++e) {
        const int m = tm * 16 + khalf * 8 + e;     // C/D layout: m = e + 8*(lane>=16)
        float v = acc[e] + bv;
        if (g.Yin)         v += g.Yin[(size_t)m * g.ldYin + ncol];
        if (g.act == 1)      v = fmaxf(v, 0.0f);
        else if (g.act == 2) v = tanhf(v);
        if (g.mlen && !(g.mlen[m] > g.tcur)) v = 0.0f;
        g.Y[(size_t)m * g.ldY + ncol] = v;
        if (g.Ybf) g.Ybf[(size_t)m * g.N + ncol] = (__bf16)v;
    }
}

// Grouped GEMM: tiles [0,tiles0) -> g0, [tiles0,tilesTot) -> g1.
__global__ void __launch_bounds__(256)
wmma_gemm_multi(GemmSet g0, GemmSet g1, int tiles0, int tilesTot)
{
    const int lane = threadIdx.x & 31;
    const int tile = blockIdx.x * 8 + (threadIdx.x >> 5);
    if (tile >= tilesTot) return;                   // wave-uniform exit
    if (tile < tiles0) gemm_tile(g0, tile, lane);
    else               gemm_tile(g1, tile - tiles0, lane);
}

// fp32-source GEMM with row remap (for the image-feature projections).
__global__ void __launch_bounds__(256)
wmma_gemm_f32src(const float* __restrict__ X, int ldX,
                 const int* __restrict__ rowmap, int rpg,
                 const bf16_t* __restrict__ W, const float* __restrict__ bias,
                 float* __restrict__ Y, int ldY, bf16_t* __restrict__ Ybf,
                 int M, int N, int K, int act)
{
    const int lane = threadIdx.x & 31;
    const int ntn  = N >> 4;
    const int tile = blockIdx.x * 8 + (threadIdx.x >> 5);
    if (tile >= (M >> 4) * ntn) return;
    const int tm = tile / ntn, tn = tile - tm * ntn;
    const int khalf = lane >> 4;
    const int mrow  = tm * 16 + (lane & 15);
    const int ncol  = tn * 16 + (lane & 15);

    const int src = rowmap[mrow / rpg] * rpg + (mrow % rpg);
    const float*  xrow = X + (size_t)src * ldX;
    const bf16_t* wrow = W + (size_t)ncol * K + khalf * 16;

    v8f acc = {};
    for (int k0 = 0; k0 < K; k0 += 32) {
        v16bf a, b;
#pragma unroll
        for (int e = 0; e < 8; ++e) {
            a[e]     = (__bf16)xrow[k0 + khalf * 16 + e];
            a[8 + e] = (__bf16)xrow[k0 + 8 + khalf * 16 + e];
        }
        b = *(const v16bf*)(wrow + k0);
        acc = __builtin_amdgcn_wmma_f32_16x16x32_bf16(false, a, false, b,
                                                      (short)0, acc, false, false);
    }
    const float bv = bias[ncol];
#pragma unroll
    for (int e = 0; e < 8; ++e) {
        const int m = tm * 16 + khalf * 8 + e;
        float v = acc[e] + bv;
        if (act == 1) v = fmaxf(v, 0.0f);
        Y[(size_t)m * ldY + ncol] = v;
        if (Ybf) Ybf[(size_t)m * N + ncol] = (__bf16)v;
    }
}

// ---------------------------------------------------------------------------
// Fused per-step attention:  WMMA (sen_att | h_att) -> LDS, scores/softmax/
// context in LDS, then ctx_W WMMA with A fragments read from LDS (ds_load).
// One block handles 16 batch rows; grid = B/16 = 8.
// ---------------------------------------------------------------------------
__global__ void __launch_bounds__(256)
attention_ctx(const bf16_t* __restrict__ senaff_bfa, const bf16_t* __restrict__ haff_bfa,
              const bf16_t* __restrict__ senattW, const float* __restrict__ senattB,
              const bf16_t* __restrict__ hattW,   const float* __restrict__ hattB,
              const float* __restrict__ vattn,    const float* __restrict__ spatial,
              const float* __restrict__ sen_aff_f,const float* __restrict__ h_aff_f,
              const float* __restrict__ alphW,    const float* __restrict__ alphB,
              const bf16_t* __restrict__ ctxW,    const float* __restrict__ ctxB,
              bf16_t* __restrict__ outl_bf)
{
    __shared__ float  s_att[2][16][HID_];   // [0]=sen_att, [1]=h_att   (64KB)
    __shared__ bf16_t s_ctxh[16][HID_];     // bf16(context + h_aff)    (16KB)
    __shared__ float  s_al[16][64];         // scores/alphas (50 used)  (4KB)

    const int b0    = blockIdx.x * 16;
    const int lane  = threadIdx.x & 31;
    const int wave  = threadIdx.x >> 5;
    const int khalf = lane >> 4;

    // phase 0: s_att[g] = [senaff|haff](b0..b0+15) @ [senattW|hattW]^T + bias
    for (int tt = wave; tt < 64; tt += 8) {
        const int g  = tt >> 5;                 // 0: sen_att, 1: h_att
        const int tn = tt & 31;
        const bf16_t* Xa = g ? haff_bfa : senaff_bfa;
        const bf16_t* Wa = g ? hattW    : senattW;
        const float*  Ba = g ? hattB    : senattB;
        const int ncol = tn * 16 + (lane & 15);
        const bf16_t* xrow = Xa + (size_t)(b0 + (lane & 15)) * HID_;
        const bf16_t* wrow = Wa + (size_t)ncol * HID_ + khalf * 16;
        v8f acc = {};
        for (int k0 = 0; k0 < HID_; k0 += 32) {
            v8bf alo = *(const v8bf*)(xrow + k0 + khalf * 16);
            v8bf ahi = *(const v8bf*)(xrow + k0 + 8 + khalf * 16);
            v16bf a, b;
#pragma unroll
            for (int e = 0; e < 8; ++e) { a[e] = alo[e]; a[8 + e] = ahi[e]; }
            b = *(const v16bf*)(wrow + k0);
            acc = __builtin_amdgcn_wmma_f32_16x16x32_bf16(false, a, false, b,
                                                          (short)0, acc, false, false);
        }
        const float bv = Ba[ncol];
#pragma unroll
        for (int e = 0; e < 8; ++e)
            s_att[g][khalf * 8 + e][ncol] = acc[e] + bv;
    }
    __syncthreads();

    // phase 1a: 16*50 attention scores, one wave per score.
    for (int idx = wave; idx < 16 * (NPIX_ + 1); idx += 8) {
        const int bb = idx / (NPIX_ + 1);
        const int p  = idx - bb * (NPIX_ + 1);
        const int b  = b0 + bb;
        const float* f = (p < NPIX_) ? &vattn[((size_t)b * NPIX_ + p) * HID_]
                                     : &s_att[0][bb][0];
        float part = 0.f;
        for (int j = lane; j < HID_; j += 32)
            part += tanhf(f[j] + s_att[1][bb][j]) * alphW[j];
        for (int off = 16; off > 0; off >>= 1) part += __shfl_down(part, off, 32);
        if (lane == 0) s_al[bb][p] = part + alphB[0];
    }
    __syncthreads();

    // phase 1b: softmax over 50, one thread per batch row.
    if (threadIdx.x < 16) {
        const int bb = threadIdx.x;
        float mx = s_al[bb][0];
        for (int p = 1; p <= NPIX_; ++p) mx = fmaxf(mx, s_al[bb][p]);
        float s = 0.f;
        for (int p = 0; p <= NPIX_; ++p) { float e = __expf(s_al[bb][p] - mx); s_al[bb][p] = e; s += e; }
        const float inv = 1.f / s;
        for (int p = 0; p <= NPIX_; ++p) s_al[bb][p] *= inv;
    }
    __syncthreads();

    // phase 1c: context + h_aff -> bf16 in LDS.
    for (int idx = threadIdx.x; idx < 16 * HID_; idx += 256) {
        const int bb = idx >> 9;
        const int j  = idx & (HID_ - 1);
        const int b  = b0 + bb;
        float ctx = s_al[bb][NPIX_] * sen_aff_f[(size_t)b * HID_ + j];
        for (int p = 0; p < NPIX_; ++p)
            ctx += s_al[bb][p] * spatial[((size_t)b * NPIX_ + p) * HID_ + j];
        s_ctxh[bb][j] = (__bf16)(ctx + h_aff_f[(size_t)b * HID_ + j]);
    }
    __syncthreads();

    // phase 2: out_l = tanh(ctxh @ ctxW^T + ctxB), A fragments from LDS.
    for (int tn = wave; tn < 32; tn += 8) {
        const int ncol = tn * 16 + (lane & 15);
        const bf16_t* xrow = &s_ctxh[lane & 15][0];
        const bf16_t* wrow = ctxW + (size_t)ncol * HID_ + khalf * 16;
        v8f acc = {};
        for (int k0 = 0; k0 < HID_; k0 += 32) {
            v8bf alo = *(const v8bf*)(xrow + k0 + khalf * 16);
            v8bf ahi = *(const v8bf*)(xrow + k0 + 8 + khalf * 16);
            v16bf a, b;
#pragma unroll
            for (int e = 0; e < 8; ++e) { a[e] = alo[e]; a[8 + e] = ahi[e]; }
            b = *(const v16bf*)(wrow + k0);
            acc = __builtin_amdgcn_wmma_f32_16x16x32_bf16(false, a, false, b,
                                                          (short)0, acc, false, false);
        }
        const float bv = ctxB[ncol];
#pragma unroll
        for (int e = 0; e < 8; ++e) {
            const float v = tanhf(acc[e] + bv);
            outl_bf[(size_t)(b0 + khalf * 8 + e) * HID_ + ncol] = (__bf16)v;
        }
    }
}

// ---------------------------------------------------------------------------
// Small kernels.
// ---------------------------------------------------------------------------
__global__ void argsort_desc(const int* __restrict__ len2d,
                             int* __restrict__ order, int* __restrict__ len_s)
{
    __shared__ int sl[B_];
    int i = threadIdx.x;
    if (i < B_) sl[i] = len2d[i];
    __syncthreads();
    if (i < B_) {
        int li = sl[i], rank = 0;
        for (int j = 0; j < B_; ++j) {
            int lj = sl[j];
            if (lj > li || (lj == li && j < i)) ++rank;
        }
        order[rank] = i;
        len_s[rank] = li;
    }
}

__global__ void f2bf(const float* __restrict__ s, bf16_t* __restrict__ d, int n)
{
    int i = blockIdx.x * 256 + threadIdx.x;
    if (i < n) d[i] = (__bf16)s[i];
}

__global__ void addvec(const float* __restrict__ a, const float* __restrict__ b,
                       float* __restrict__ o, int n)
{
    int i = blockIdx.x * 256 + threadIdx.x;
    if (i < n) o[i] = a[i] + b[i];
}

__global__ void init_state(float* __restrict__ h, float* __restrict__ c,
                           bf16_t* __restrict__ hbf)
{
    int i = blockIdx.x * 256 + threadIdx.x;
    if (i < B_ * HID_) { h[i] = 0.f; c[i] = 0.f; hbf[i] = (__bf16)0.f; }
}

// inputs[b,t,:] = concat( bf16(emb_W[cap[order[b],t]]), bf16(gimg[b]) )
__global__ void build_inputs(const int* __restrict__ caps, const int* __restrict__ order,
                             const float* __restrict__ embW, const float* __restrict__ gimg,
                             bf16_t* __restrict__ inputs)
{
    int bt = blockIdx.x;
    int b = bt / T_, t = bt - b * T_;
    int tok = caps[order[b] * T_ + t];
    bf16_t* dst = inputs + (size_t)bt * (2 * EMB_);
    const float* e = embW + (size_t)tok * EMB_;
    const float* g = gimg + (size_t)b * EMB_;
    for (int j = threadIdx.x; j < EMB_; j += blockDim.x) {
        dst[j]        = (__bf16)e[j];
        dst[EMB_ + j] = (__bf16)g[j];
    }
}

// LSTM pointwise + sentinel gate + masked carry commit; gates row = 2560 wide.
__global__ void lstm_step(const float* __restrict__ gates,
                          float* __restrict__ hprev, float* __restrict__ cprev,
                          bf16_t* __restrict__ hprev_bf,
                          bf16_t* __restrict__ hnew_bf, bf16_t* __restrict__ st_bf,
                          const int* __restrict__ len_s, int t)
{
    int idx = blockIdx.x * 256 + threadIdx.x;
    if (idx >= B_ * HID_) return;
    int b = idx >> 9;
    int j = idx & (HID_ - 1);
    const float* g = gates + (size_t)b * NGATE_;
    float ig = sigmoidf_(g[j]);
    float fg = sigmoidf_(g[HID_ + j]);
    float gg = tanhf(g[2 * HID_ + j]);
    float og = sigmoidf_(g[3 * HID_ + j]);
    float sv = g[4 * HID_ + j];                 // sentinel pre-activation
    float cn = fg * cprev[idx] + ig * gg;
    float hn = og * tanhf(cn);
    float st = sigmoidf_(sv) * tanhf(cn);
    hnew_bf[idx] = (__bf16)hn;                  // unmasked h_new feeds this step
    st_bf[idx]   = (__bf16)st;
    bool m = len_s[b] > t;                      // mask affects only the carry
    float hp = m ? hn : hprev[idx];
    hprev[idx] = hp;
    hprev_bf[idx] = (__bf16)hp;
    cprev[idx] = m ? cn : cprev[idx];
}

// ---------------------------------------------------------------------------
extern "C" void kernel_launch(void* const* d_in, const int* in_sizes, int n_in,
                              void* d_out, int out_size, void* d_ws, size_t ws_size,
                              hipStream_t stream)
{
    (void)in_sizes; (void)n_in; (void)out_size; (void)ws_size;
    const float* enc_image = (const float*)d_in[0];
    const float* gfeat     = (const float*)d_in[1];
    const int*   caps      = (const int*)d_in[2];
    const int*   caplen    = (const int*)d_in[3];
    const float* embW      = (const float*)d_in[4];
    const float* e2hW = (const float*)d_in[5];  const float* e2hB = (const float*)d_in[6];
    const float* gfW  = (const float*)d_in[7];  const float* gfB  = (const float*)d_in[8];
    const float* fcW  = (const float*)d_in[9];  const float* fcB  = (const float*)d_in[10];
    const float* WihW = (const float*)d_in[11]; const float* bih  = (const float*)d_in[12];
    const float* WhhW = (const float*)d_in[13]; const float* bhh  = (const float*)d_in[14];
    const float* xgW  = (const float*)d_in[15]; const float* xgB  = (const float*)d_in[16];
    const float* hgW  = (const float*)d_in[17]; const float* hgB  = (const float*)d_in[18];
    const float* senaffW = (const float*)d_in[19]; const float* senaffB = (const float*)d_in[20];
    const float* senattW = (const float*)d_in[21]; const float* senattB = (const float*)d_in[22];
    const float* haffW   = (const float*)d_in[23]; const float* haffB   = (const float*)d_in[24];
    const float* hattW   = (const float*)d_in[25]; const float* hattB   = (const float*)d_in[26];
    const float* vattW   = (const float*)d_in[27]; const float* vattB   = (const float*)d_in[28];
    const float* alphW   = (const float*)d_in[29]; const float* alphB   = (const float*)d_in[30];
    const float* ctxW    = (const float*)d_in[31]; const float* ctxB    = (const float*)d_in[32];
    float* out = (float*)d_out;

    char* p = (char*)d_ws;
    auto alloc = [&](size_t bytes) -> void* {
        void* r = (void*)p; p += (bytes + 255) & ~(size_t)255; return r;
    };

    int* order = (int*)alloc(B_ * 4);
    int* len_s = (int*)alloc(B_ * 4);

    bf16_t* e2h_bf     = (bf16_t*)alloc((size_t)HID_ * ENC_ * 2);
    bf16_t* gf_bf      = (bf16_t*)alloc((size_t)EMB_ * ENC_ * 2);
    bf16_t* fc_bf      = (bf16_t*)alloc((size_t)VOCAB_ * HID_ * 2);
    bf16_t* Wx_cat     = (bf16_t*)alloc((size_t)NGATE_ * 2 * EMB_ * 2);  // [2560,1024]
    bf16_t* Wh_cat     = (bf16_t*)alloc((size_t)NGATE_ * HID_ * 2);     // [2560,512]
    bf16_t* senaffW_bf = (bf16_t*)alloc((size_t)HID_ * HID_ * 2);
    bf16_t* senattW_bf = (bf16_t*)alloc((size_t)ATT_ * HID_ * 2);
    bf16_t* haffW_bf   = (bf16_t*)alloc((size_t)HID_ * HID_ * 2);
    bf16_t* hattW_bf   = (bf16_t*)alloc((size_t)ATT_ * HID_ * 2);
    bf16_t* vattW_bf   = (bf16_t*)alloc((size_t)ATT_ * HID_ * 2);
    bf16_t* ctxW_bf    = (bf16_t*)alloc((size_t)HID_ * HID_ * 2);

    float* bsum_cat = (float*)alloc(NGATE_ * 4);

    float*  spatial_f  = (float*)alloc((size_t)B_ * NPIX_ * HID_ * 4);
    bf16_t* spatial_bf = (bf16_t*)alloc((size_t)B_ * NPIX_ * HID_ * 2);
    float*  vattn_f    = (float*)alloc((size_t)B_ * NPIX_ * ATT_ * 4);
    float*  gimg_f     = (float*)alloc((size_t)B_ * EMB_ * 4);
    bf16_t* inputs_bf  = (bf16_t*)alloc((size_t)B_ * T_ * 2 * EMB_ * 2);
    float*  xpre_f     = (float*)alloc((size_t)B_ * T_ * NGATE_ * 4);   // precomputed xt-side gates

    float*  gates_f   = (float*)alloc((size_t)B_ * NGATE_ * 4);
    float*  hprev_f   = (float*)alloc((size_t)B_ * HID_ * 4);
    float*  cprev_f   = (float*)alloc((size_t)B_ * HID_ * 4);
    bf16_t* hprev_bf  = (bf16_t*)alloc((size_t)B_ * HID_ * 2);
    bf16_t* hnew_bf   = (bf16_t*)alloc((size_t)B_ * HID_ * 2);
    bf16_t* st_bf     = (bf16_t*)alloc((size_t)B_ * HID_ * 2);
    float*  senaff_f  = (float*)alloc((size_t)B_ * HID_ * 4);
    bf16_t* senaff_bfa= (bf16_t*)alloc((size_t)B_ * HID_ * 2);
    float*  haff_f    = (float*)alloc((size_t)B_ * HID_ * 4);
    bf16_t* haff_bfa  = (bf16_t*)alloc((size_t)B_ * HID_ * 2);
    bf16_t* outl_bf   = (bf16_t*)alloc((size_t)B_ * HID_ * 2);

    auto cvt = [&](const float* s, bf16_t* d, size_t n) {
        f2bf<<<(unsigned)((n + 255) / 256), 256, 0, stream>>>(s, d, (int)n);
    };
    auto launch1 = [&](const GemmSet& g, int M) {
        int tiles = (M / 16) * (g.N / 16);
        wmma_gemm_multi<<<(tiles + 7) / 8, 256, 0, stream>>>(g, g, tiles, tiles);
    };
    auto launch2 = [&](const GemmSet& g0, int M0, const GemmSet& g1, int M1) {
        int t0 = (M0 / 16) * (g0.N / 16);
        int tt = t0 + (M1 / 16) * (g1.N / 16);
        wmma_gemm_multi<<<(tt + 7) / 8, 256, 0, stream>>>(g0, g1, t0, tt);
    };

    // ---------------- precompute ----------------
    argsort_desc<<<1, 128, 0, stream>>>(caplen, order, len_s);

    cvt(e2hW,    e2h_bf,     (size_t)HID_ * ENC_);
    cvt(gfW,     gf_bf,      (size_t)EMB_ * ENC_);
    cvt(fcW,     fc_bf,      (size_t)VOCAB_ * HID_);
    cvt(WihW,    Wx_cat,                      (size_t)4 * HID_ * 2 * EMB_);
    cvt(xgW,     Wx_cat + (size_t)4 * HID_ * 2 * EMB_, (size_t)HID_ * 2 * EMB_);
    cvt(WhhW,    Wh_cat,                      (size_t)4 * HID_ * HID_);
    cvt(hgW,     Wh_cat + (size_t)4 * HID_ * HID_,     (size_t)HID_ * HID_);
    cvt(senaffW, senaffW_bf, (size_t)HID_ * HID_);
    cvt(senattW, senattW_bf, (size_t)ATT_ * HID_);
    cvt(haffW,   haffW_bf,   (size_t)HID_ * HID_);
    cvt(hattW,   hattW_bf,   (size_t)ATT_ * HID_);
    cvt(vattW,   vattW_bf,   (size_t)ATT_ * HID_);
    cvt(ctxW,    ctxW_bf,    (size_t)HID_ * HID_);

    addvec<<<(4 * HID_ + 255) / 256, 256, 0, stream>>>(bih, bhh, bsum_cat, 4 * HID_);
    addvec<<<(HID_ + 255) / 256, 256, 0, stream>>>(xgB, hgB, bsum_cat + 4 * HID_, HID_);
    init_state<<<(B_ * HID_ + 255) / 256, 256, 0, stream>>>(hprev_f, cprev_f, hprev_bf);

    // spatial = relu(enc_image[order] @ e2h^T + b);  gimg = relu(gfeat[order] @ gf^T + b)
    {
        int tiles = (B_ * NPIX_ / 16) * (HID_ / 16);
        wmma_gemm_f32src<<<(tiles + 7) / 8, 256, 0, stream>>>(
            enc_image, ENC_, order, NPIX_, e2h_bf, e2hB,
            spatial_f, HID_, spatial_bf, B_ * NPIX_, HID_, ENC_, 1);
        tiles = (B_ / 16) * (EMB_ / 16);
        wmma_gemm_f32src<<<(tiles + 7) / 8, 256, 0, stream>>>(
            gfeat, ENC_, order, 1, gf_bf, gfB,
            gimg_f, EMB_, nullptr, B_, EMB_, ENC_, 1);
    }
    build_inputs<<<B_ * T_, 256, 0, stream>>>(caps, order, embW, gimg_f, inputs_bf);

    // vattn = spatial @ vatt^T + vatt_b
    {
        GemmSet g = { spatial_bf, HID_, vattW_bf, vattB, vattn_f, ATT_, nullptr,
                      nullptr, 0, ATT_, HID_, 0, nullptr, 0 };
        launch1(g, B_ * NPIX_);
    }
    // xpre[b,t,:] = inputs @ Wx_cat^T + (bih+bhh | xg_b+hg_b)   — one big GEMM
    {
        GemmSet g = { inputs_bf, 2 * EMB_, Wx_cat, bsum_cat, xpre_f, NGATE_, nullptr,
                      nullptr, 0, NGATE_, 2 * EMB_, 0, nullptr, 0 };
        launch1(g, B_ * T_);
    }

    // ---------------- 30-step scan: 5 launches/step ----------------
    for (int t = 0; t < T_; ++t) {
        // 1. gates = xpre[:,t,:] + h_prev @ Wh_cat^T
        GemmSet gh = { hprev_bf, HID_, Wh_cat, nullptr, gates_f, NGATE_, nullptr,
                       xpre_f + (size_t)t * NGATE_, T_ * NGATE_, NGATE_, HID_, 0, nullptr, 0 };
        launch1(gh, B_);

        // 2. LSTM pointwise + sentinel gate + masked carry commit
        lstm_step<<<(B_ * HID_ + 255) / 256, 256, 0, stream>>>(
            gates_f, hprev_f, cprev_f, hprev_bf, hnew_bf, st_bf, len_s, t);

        // 3. grouped: sen_aff = relu(st@senaffW^T+b)  |  h_aff = tanh(h_new@haffW^T+b)
        GemmSet ga = { st_bf, HID_, senaffW_bf, senaffB, senaff_f, HID_, senaff_bfa,
                       nullptr, 0, HID_, HID_, 1, nullptr, 0 };
        GemmSet gb = { hnew_bf, HID_, haffW_bf, haffB, haff_f, HID_, haff_bfa,
                       nullptr, 0, HID_, HID_, 2, nullptr, 0 };
        launch2(ga, B_, gb, B_);

        // 4. fused: sen_att/h_att WMMA -> LDS, scores/softmax/context, ctx GEMM from LDS
        attention_ctx<<<B_ / 16, 256, 0, stream>>>(
            senaff_bfa, haff_bfa, senattW_bf, senattB, hattW_bf, hattB,
            vattn_f, spatial_f, senaff_f, haff_f, alphW, alphB,
            ctxW_bf, ctxB, outl_bf);

        // 5. pt = out_l @ fc^T + fc_b (masked) straight into d_out[:, t, :]
        GemmSet gf = { outl_bf, HID_, fc_bf, fcB, out + (size_t)t * VOCAB_, T_ * VOCAB_,
                       nullptr, nullptr, 0, VOCAB_, HID_, 0, len_s, t };
        launch1(gf, B_);
    }
}
